// MentionEncoder_59803124629553
// MI455X (gfx1250) — matrix operand order repaired and checked by reading, not compiled, not run on hardware
//
#include <hip/hip_runtime.h>
#include <hip/hip_bf16.h>

// Problem constants
#define EOUT 768
#define DIN  2304
#define HID  2304
#define G4   9216          // 4*H
#define BB   16
#define TT   64
#define KC   32            // WMMA K chunk (bf16)
#define NKC  (DIN / KC)    // 72 K-chunks
#define NKH  (NKC / 2)     // 36 per K-half in the step kernel

typedef __attribute__((ext_vector_type(16))) __bf16 v16bf;
typedef __attribute__((ext_vector_type(8)))  float  v8f;

union Frag { v16bf v; uint4 q[2]; unsigned int u[8]; };

__device__ __forceinline__ unsigned short f2bf(float f) {
    unsigned u = __builtin_bit_cast(unsigned, f);
    u += 0x7FFFu + ((u >> 16) & 1u);          // round-to-nearest-even
    return (unsigned short)(u >> 16);
}

// ---------------- fp32 -> bf16 weight conversion (once per call) ------------
// float4 in -> 4 packed bf16 (uint2) out; n must be a multiple of 4 (it is).
__global__ void cvt_f32_bf16_v4(const float4* __restrict__ src,
                                uint2* __restrict__ dst, int n4) {
    int i = blockIdx.x * blockDim.x + threadIdx.x;
    int stride = gridDim.x * blockDim.x;
    for (; i < n4; i += stride) {
        float4 f = src[i];
        uint2 o;
        o.x = (unsigned)f2bf(f.x) | ((unsigned)f2bf(f.y) << 16);
        o.y = (unsigned)f2bf(f.z) | ((unsigned)f2bf(f.w) << 16);
        dst[i] = o;
    }
}

__global__ void zero_f32(float* p, int n) {
    int i = blockIdx.x * blockDim.x + threadIdx.x;
    if (i < n) p[i] = 0.f;
}
__global__ void zero_u16(unsigned short* p, int n) {
    int i = blockIdx.x * blockDim.x + threadIdx.x;
    if (i < n) p[i] = 0;
}

// ---------------- sequence lengths ------------------------------------------
__global__ void lens_kernel(const int* __restrict__ mask, const int* __restrict__ start,
                            const int* __restrict__ end,
                            int* __restrict__ lenRaw, int* __restrict__ lenClamp) {
    int b = threadIdx.x;
    if (b < BB) {
        int length = 0;
        for (int t = 0; t < TT; ++t) length += mask[b * TT + t];
        int l0 = start[b] - 1;          // left valid count
        int l1 = length - end[b];       // right valid count
        lenRaw[b]        = l0;
        lenRaw[BB + b]   = l1;
        lenClamp[b]      = l0 > 1 ? l0 : 1;
        lenClamp[BB + b] = l1 > 1 ? l1 : 1;
    }
}

// ---------------- build left/right sequences in bf16 ------------------------
// layout: seq[s][b][t][d], flat row R = s*1024 + b*64 + t
__global__ void seq_kernel(const float* __restrict__ mf, const int* __restrict__ end,
                           const int* __restrict__ lenRaw,
                           unsigned short* __restrict__ seq) {
    int e = blockIdx.x * blockDim.x + threadIdx.x;
    if (e >= 2 * BB * TT * DIN) return;
    int d = e % DIN;
    int t = (e / DIN) % TT;
    int b = (e / (DIN * TT)) % BB;
    int s = e / (DIN * TT * BB);
    float v = 0.f;
    if (s == 0) {
        if (t < TT - 1 && t < lenRaw[b]) v = mf[(b * TT + t + 1) * DIN + d];
    } else {
        if (t < lenRaw[BB + b]) {
            int idx = end[b] + t;
            idx = idx < 0 ? 0 : (idx > TT - 1 ? TT - 1 : idx);
            v = mf[(b * TT + idx) * DIN + d];
        }
    }
    seq[e] = f2bf(v);
}

// ---------------- xp = seq @ W_ih^T + (b_ih + b_hh)  (big parallel GEMM) ----
// 2048x9216 = [2048x2304]x[2304x9216], bf16 WMMA, fp32 accumulate.
// Block: 128 threads (4 waves). Macro tile 64(M) x 64(N); wave w owns M-tile w
// and sweeps the 4 N-tiles. LDS tiles stored [row][16 dwords], row stride 20
// dwords (16B-aligned + conflict-free) so fragments load as ds_load_b128.
__global__ __launch_bounds__(128) void xp_gemm(const unsigned int* __restrict__ seq_bf,
                                               const unsigned int* __restrict__ wih_bf,
                                               const float* __restrict__ b_ih,
                                               const float* __restrict__ b_hh,
                                               float* __restrict__ xp) {
    __shared__ unsigned int As[64 * 20];
    __shared__ unsigned int Bs[64 * 20];
    const int tid  = threadIdx.x;
    const int lane = tid & 31;
    const int w    = tid >> 5;
    const int hi   = (lane >> 4) & 1;
    const int ln   = lane & 15;
    const int m0   = blockIdx.x * 64;
    const int n0   = blockIdx.y * 64;

    const uint4* seq4 = (const uint4*)seq_bf;
    const uint4* wih4 = (const uint4*)wih_bf;

    v8f acc[4] = {v8f{}, v8f{}, v8f{}, v8f{}};

    for (int kc = 0; kc < NKC; ++kc) {
        const int d0w = kc * 16;  // dword K base
        // stage 64x32 bf16 A and B tiles: 256 uint4 each, 2 per thread
        #pragma unroll
        for (int i = 0; i < 2; ++i) {
            int e4  = i * 128 + tid;
            int row = e4 >> 2;
            int grp = e4 & 3;
            *(uint4*)&As[row * 20 + grp * 4] = seq4[((m0 + row) * 1152 + d0w) / 4 + grp];
            *(uint4*)&Bs[row * 20 + grp * 4] = wih4[((n0 + row) * 1152 + d0w) / 4 + grp];
        }
        __syncthreads();

        Frag af, bfr[4];
        const int arow = w * 16 + ln;
        af.q[0] = *(const uint4*)&As[arow * 20 + hi * 4];       // K 0..7 / 8..15
        af.q[1] = *(const uint4*)&As[arow * 20 + 8 + hi * 4];   // K 16..23 / 24..31
        #pragma unroll
        for (int nt = 0; nt < 4; ++nt) {
            bfr[nt].q[0] = *(const uint4*)&Bs[(nt * 16 + ln) * 20 + hi * 4];
            bfr[nt].q[1] = *(const uint4*)&Bs[(nt * 16 + ln) * 20 + hi * 4 + 8];
        }
        #pragma unroll
        for (int nt = 0; nt < 4; ++nt)
            acc[nt] = __builtin_amdgcn_wmma_f32_16x16x32_bf16(
                false, af.v, false, bfr[nt].v, (short)0, acc[nt], false, false);
        __syncthreads();
    }

    #pragma unroll
    for (int nt = 0; nt < 4; ++nt) {
        int col   = n0 + nt * 16 + ln;
        float bsum = b_ih[col] + b_hh[col];
        #pragma unroll
        for (int r = 0; r < 8; ++r) {
            int R = m0 + w * 16 + r + 8 * hi;
            xp[R * G4 + col] = acc[nt][r] + bsum;
        }
    }
}

// ---------------- one LSTM timestep (both sequences) ------------------------
// grid.x = 2 * (H/16) = 288 blocks; block = 256 threads (8 waves).
// Wave w: gate = w&3 (i,f,g,o), K-half = w>>2. Each wave runs a 36-deep WMMA
// chain (K split halves the serial dependency); halves combine in LDS during
// the fused sigmoid/tanh cell update. All global fragment reads are b128.
__global__ __launch_bounds__(256) void step_kernel(const float* __restrict__ xp,
        const unsigned int* __restrict__ hbf_in, unsigned short* __restrict__ hbf_out,
        float* __restrict__ hstate, float* __restrict__ cstate,
        const unsigned int* __restrict__ whh_bf, const int* __restrict__ lenClamp,
        int t) {
    __shared__ float gl[8 * 16 * 16];
    const int tid  = threadIdx.x;
    const int lane = tid & 31;
    const int w    = tid >> 5;           // 0..7
    const int gate = w & 3;
    const int kh   = w >> 2;             // K half
    const int hi   = (lane >> 4) & 1;
    const int ln   = lane & 15;
    const int s    = blockIdx.x / (HID / 16);
    const int jc   = blockIdx.x % (HID / 16);
    const int j0   = jc * 16;
    const int gbase = gate * HID + j0;   // gate-row base within [0,4H)

    const uint4* h4 = (const uint4*)hbf_in;
    const uint4* w4 = (const uint4*)whh_bf;

    v8f acc = {};
    const int arowbase = (s * BB + ln) * 1152;   // h_bf dword row (batch = ln)
    const int brow     = (gbase + ln) * 1152;    // W_hh dword row
    for (int kc = kh * NKH; kc < kh * NKH + NKH; ++kc) {
        const int k0w = kc * 16;
        Frag af, bf;
        af.q[0] = h4[(arowbase + k0w + hi * 4) >> 2];
        af.q[1] = h4[(arowbase + k0w + hi * 4 + 8) >> 2];
        bf.q[0] = w4[(brow + k0w + hi * 4) >> 2];
        bf.q[1] = w4[(brow + k0w + hi * 4 + 8) >> 2];
        acc = __builtin_amdgcn_wmma_f32_16x16x32_bf16(
            false, af.v, false, bf.v, (short)0, acc, false, false);
    }

    // partial gates -> LDS; K-half 0 also folds in the precomputed xp term
    const int xcol = gbase + ln;
    #pragma unroll
    for (int r = 0; r < 8; ++r) {
        int m  = r + 8 * hi;                     // batch
        float v = acc[r];
        if (kh == 0) {
            int xr = s * 1024 + m * 64 + t;      // xp row
            v += xp[xr * G4 + xcol];
        }
        gl[w * 256 + m * 16 + ln] = v;
    }
    __syncthreads();

    // fused cell update: 256 (b,j) elements, 1 per thread
    {
        int e  = tid;
        int b  = e >> 4;
        int jj = e & 15;
        int sb = s * BB + b;
        int idx = sb * HID + (j0 + jj);
        float hCur;
        if (t < lenClamp[sb]) {
            int x = b * 16 + jj;
            float gi = gl[0 * 256 + x] + gl[4 * 256 + x];
            float gf = gl[1 * 256 + x] + gl[5 * 256 + x];
            float gg = gl[2 * 256 + x] + gl[6 * 256 + x];
            float go = gl[3 * 256 + x] + gl[7 * 256 + x];
            float si = 1.f / (1.f + __expf(-gi));
            float sf = 1.f / (1.f + __expf(-gf));
            float so = 1.f / (1.f + __expf(-go));
            float cn = sf * cstate[idx] + si * tanhf(gg);
            float hn = so * tanhf(cn);
            cstate[idx] = cn;
            hstate[idx] = hn;
            hCur = hn;
        } else {
            hCur = hstate[idx];                  // frozen state
        }
        hbf_out[idx] = f2bf(hCur);
    }
}

// ---------------- out = [hL, hR] @ W_out^T + b_out  (tiny, fp32 VALU) -------
__global__ void out_kernel(const float* __restrict__ h, const float* __restrict__ W_out,
                           const float* __restrict__ b_out, float* __restrict__ out) {
    int i = blockIdx.x * blockDim.x + threadIdx.x;
    if (i >= BB * EOUT) return;
    int b = i / EOUT, e = i % EOUT;
    const float* hL = h + b * HID;
    const float* hR = h + (BB + b) * HID;
    const float* wr = W_out + e * (2 * HID);
    float acc = b_out[e];
    #pragma unroll 4
    for (int k = 0; k < HID; ++k) acc += hL[k] * wr[k] + hR[k] * wr[HID + k];
    out[i] = acc;
}

extern "C" void kernel_launch(void* const* d_in, const int* in_sizes, int n_in,
                              void* d_out, int out_size, void* d_ws, size_t ws_size,
                              hipStream_t stream) {
    const float* mf    = (const float*)d_in[0];
    const int*   mask  = (const int*)  d_in[1];
    const int*   start = (const int*)  d_in[2];
    const int*   end   = (const int*)  d_in[3];
    const float* W_ih  = (const float*)d_in[4];
    const float* W_hh  = (const float*)d_in[5];
    const float* b_ih  = (const float*)d_in[6];
    const float* b_hh  = (const float*)d_in[7];
    const float* W_out = (const float*)d_in[8];
    const float* b_out = (const float*)d_in[9];
    float* out = (float*)d_out;

    char* ws = (char*)d_ws;
    constexpr size_t SZ_W    = (size_t)G4 * DIN * 2;           // 42,467,328
    constexpr size_t OFF_WIH = 0;
    constexpr size_t OFF_WHH = OFF_WIH + SZ_W;
    constexpr size_t OFF_SEQ = OFF_WHH + SZ_W;                 // 2*16*64*2304 bf16
    constexpr size_t OFF_XP  = OFF_SEQ + (size_t)2 * BB * TT * DIN * 2;
    constexpr size_t OFF_H   = OFF_XP  + (size_t)2 * BB * TT * G4 * 4;
    constexpr size_t OFF_C   = OFF_H   + (size_t)2 * BB * HID * 4;
    constexpr size_t OFF_HBF = OFF_C   + (size_t)2 * BB * HID * 4; // 2 ping-pong bufs
    constexpr size_t OFF_LEN = OFF_HBF + (size_t)2 * 2 * BB * HID * 2;

    unsigned short* wih_bf = (unsigned short*)(ws + OFF_WIH);
    unsigned short* whh_bf = (unsigned short*)(ws + OFF_WHH);
    unsigned short* seq_bf = (unsigned short*)(ws + OFF_SEQ);
    float*          xp     = (float*)(ws + OFF_XP);
    float*          hstate = (float*)(ws + OFF_H);
    float*          cstate = (float*)(ws + OFF_C);
    unsigned short* hbf    = (unsigned short*)(ws + OFF_HBF);
    int*            lenClamp = (int*)(ws + OFF_LEN);
    int*            lenRaw   = lenClamp + 2 * BB;

    // 1) bf16 weight copies (read fp32 once from HBM; later passes hit L2 at half size)
    cvt_f32_bf16_v4<<<2048, 256, 0, stream>>>((const float4*)W_ih, (uint2*)wih_bf,
                                              (G4 * DIN) / 4);
    cvt_f32_bf16_v4<<<2048, 256, 0, stream>>>((const float4*)W_hh, (uint2*)whh_bf,
                                              (G4 * HID) / 4);

    // 2) lengths + gathered/masked left & right sequences in bf16
    lens_kernel<<<1, 32, 0, stream>>>(mask, start, end, lenRaw, lenClamp);
    seq_kernel<<<(2 * BB * TT * DIN + 255) / 256, 256, 0, stream>>>(mf, end, lenRaw, seq_bf);

    // 3) zero-init LSTM state (h, c contiguous) and both h-bf16 ping-pong buffers
    zero_f32<<<(4 * BB * HID + 255) / 256, 256, 0, stream>>>(hstate, 4 * BB * HID);
    zero_u16<<<(4 * BB * HID + 255) / 256, 256, 0, stream>>>(hbf, 4 * BB * HID);

    // 4) big parallel GEMM: xp[2048 x 9216] (biases folded)
    xp_gemm<<<dim3(32, 144), 128, 0, stream>>>((const unsigned int*)seq_bf,
                                               (const unsigned int*)wih_bf,
                                               b_ih, b_hh, xp);

    // 5) 64 serialized recurrent steps, both LSTMs per launch, ping-pong h
    for (int t = 0; t < TT; ++t) {
        const unsigned int* hin = (const unsigned int*)(hbf + (size_t)(t & 1) * 2 * BB * HID);
        unsigned short* hout    = hbf + (size_t)((t + 1) & 1) * 2 * BB * HID;
        step_kernel<<<2 * (HID / 16), 256, 0, stream>>>(xp, hin, hout, hstate, cstate,
                                                        (const unsigned int*)whh_bf,
                                                        lenClamp, t);
    }

    // 6) tiny output projection in fp32
    out_kernel<<<(BB * EOUT + 255) / 256, 256, 0, stream>>>(hstate, W_out, b_out, out);

    (void)in_sizes; (void)n_in; (void)out_size; (void)ws_size;
}
